// IrrepsToHessian_56607668961468
// MI455X (gfx1250) — matrix operand order repaired and checked by reading, not compiled, not run on hardware
//
#include <hip/hip_runtime.h>
#include <hip/hip_bf16.h>

typedef __attribute__((ext_vector_type(16))) _Float16 v16h;
typedef __attribute__((ext_vector_type(8)))  _Float16 v8h;
typedef __attribute__((ext_vector_type(8)))  float    v8f;

#define NMUL    16
#define NNODES  4096
#define NATOM   64
#define NPATH   15
#define NPAIRS  (NNODES * NATOM)   // 262144
#define KPAD    160                // 144 features padded to 5 * 32
#define HROWS   16                 // 9 irrep->cart rows padded to WMMA M
#define CSTRIDE 16                 // padded per-pair stride of cart scratch

// path tables: (l1, l2, lo) for the 15 CG paths, in reference order
__constant__ int dPL1[NPATH] = {0,1,2,0,1,1,1,2,2,0,2,1,1,2,2};
__constant__ int dPL2[NPATH] = {0,1,2,1,0,1,2,1,2,2,0,1,2,1,2};
__constant__ int dPLO[NPATH] = {0,0,0,1,1,1,1,1,1,2,2,2,2,2,2};
// prefix offsets of w3j tensors (sizes (2l1+1)(2l2+1)(2lo+1)), total 615 floats
__constant__ int dW3OFF[NPATH+1] = {0,1,10,35,44,53,80,125,170,245,270,295,340,415,490,615};
// prefix offsets of mixed features t_p (16*(2l1+1) each), total 816 floats
__constant__ int dTOFF[NPATH+1]  = {0,16,64,144,160,208,256,304,384,464,480,560,608,656,736,816};
__constant__ int dXOFF[3] = {0,16,64};   // feature offset of l-block
__constant__ int dKOFF[3] = {0,1,4};     // irrep output offset of l-block
// alpha = sqrt(2lo+1)/sqrt(FAN), FAN = {768,1536,1536}
__constant__ float dALPHA[3] = {0.03608439182435161f, 0.04419417382415922f, 0.05705445799165322f};

// ---------------- stage 1: real Wigner-3j tables (e3nn convention) -----------
struct cx { double re, im; };
__device__ inline cx cmul(cx a, cx b){ return {a.re*b.re - a.im*b.im, a.re*b.im + a.im*b.re}; }

__device__ inline double fct(int n){
  const double f[9] = {1.,1.,2.,6.,24.,120.,720.,5040.,40320.};
  return f[n];
}
__device__ inline int imax2(int a,int b){ return a>b?a:b; }
__device__ inline int imin2(int a,int b){ return a<b?a:b; }

__device__ double su2_cg(int j1,int m1,int j2,int m2,int j3,int m3){
  if (m3 != m1 + m2) return 0.0;
  int vmin = imax2(imax2(-j1 + j2 + m3, -j1 + m1), 0);
  int vmax = imin2(imin2(j2 + j3 + m1, j3 - j1 + j2), j3 + m3);
  double C = sqrt((2.0*j3 + 1.0) * fct(j3+j1-j2) * fct(j3-j1+j2) * fct(j1+j2-j3)
                  * fct(j3+m3) * fct(j3-m3)
                  / (fct(j1+j2+j3+1) * fct(j1-m1) * fct(j1+m1) * fct(j2-m2) * fct(j2+m2)));
  double S = 0.0;
  for (int v = vmin; v <= vmax; ++v){
    double sgn = ((v + j2 + m2) & 1) ? -1.0 : 1.0;
    S += sgn * fct(j2+j3+m1-v) * fct(j1-m1+v)
         / (fct(v) * fct(j3-j1+j2-v) * fct(j3+m3-v) * fct(v+j1-j2-m3));
  }
  return C * S;
}

// element (row, col) of (-i)^l * q_real_to_complex(l)
__device__ cx qelem(int l, int row, int col){
  const double is2 = 0.70710678118654752440;
  int m = row - l;
  cx q = {0., 0.};
  if (m < 0){
    if      (col == l - m) q = { is2, 0. };
    else if (col == l + m) q = { 0., -is2 };
  } else if (m == 0){
    if (col == l) q = { 1., 0. };
  } else {
    double s = (m & 1) ? -1.0 : 1.0;
    if      (col == l + m) q = { s*is2, 0. };
    else if (col == l - m) q = { 0., s*is2 };
  }
  cx ph;
  switch (l & 3){
    case 0: ph = { 1., 0.}; break;
    case 1: ph = { 0.,-1.}; break;
    case 2: ph = {-1., 0.}; break;
    default:ph = { 0., 1.}; break;
  }
  return cmul(ph, q);
}

__global__ void w3j_kernel(float* __restrict__ w3out){
  __shared__ double tmp[615];
  __shared__ double nrm[NPATH];
  const int tid = threadIdx.x;
  for (int idx = tid; idx < 615; idx += blockDim.x){
    int p = 0; while (idx >= dW3OFF[p+1]) ++p;
    const int l1 = dPL1[p], l2 = dPL2[p], l3 = dPLO[p];
    const int d1 = 2*l1+1, d2 = 2*l2+1, d3 = 2*l3+1;
    int local = idx - dW3OFF[p];
    int a = local / (d2*d3);
    int rem = local % (d2*d3);
    int b = rem / d3, c = rem % d3;
    cx acc = {0., 0.};
    for (int i = 0; i < d1; ++i){
      cx q1 = qelem(l1, i, a);
      if (q1.re == 0. && q1.im == 0.) continue;
      for (int k = 0; k < d2; ++k){
        cx q2 = qelem(l2, k, b);
        if (q2.re == 0. && q2.im == 0.) continue;
        int mi = (i - l1) + (k - l2) + l3;          // m-index of l3 fixed by m1+m2
        if (mi < 0 || mi >= d3) continue;
        double cg = su2_cg(l1, i-l1, l2, k-l2, l3, (i-l1)+(k-l2));
        if (cg == 0.) continue;
        cx q3 = qelem(l3, mi, c);
        cx q3c = { q3.re, -q3.im };
        cx t = cmul(cmul(q1, q2), q3c);
        acc.re += t.re * cg;
        acc.im += t.im * cg;
      }
    }
    tmp[idx] = acc.re;                              // imag part ~0 by construction
  }
  __syncthreads();
  if (tid < NPATH){
    double s = 0.;
    for (int i = dW3OFF[tid]; i < dW3OFF[tid+1]; ++i) s += tmp[i]*tmp[i];
    nrm[tid] = sqrt(s);
  }
  __syncthreads();
  for (int idx = tid; idx < 615; idx += blockDim.x){
    int p = 0; while (idx >= dW3OFF[p+1]) ++p;
    w3out[idx] = (float)(tmp[idx] / nrm[p]);
  }
}

// ---------------- stage 2: per-node fused operator H_n (16x160 f16) ----------
__global__ void build_h_kernel(const float* __restrict__ node_feats,
                               const float* __restrict__ tp_weights,
                               const float* __restrict__ w3,
                               _Float16* __restrict__ Hbuf,
                               _Float16* __restrict__ Xbuf){
  __shared__ float xs[144];
  __shared__ float w3s[615];
  __shared__ float wls[NPATH * 256];
  __shared__ float ts[816];
  const int n = blockIdx.x;
  const int tid = threadIdx.x;

  if (tid < 144) xs[tid] = node_feats[(size_t)n*144 + tid];
  for (int i = tid; i < 615; i += blockDim.x) w3s[i] = w3[i];
  for (int i = tid; i < NPATH*256; i += blockDim.x) wls[i] = tp_weights[i];
  __syncthreads();

  // t_p[v,i] = sum_u x[l1][u,i] * W_p[u,v]
  for (int e = tid; e < 816; e += blockDim.x){
    int p = 0; while (e >= dTOFF[p+1]) ++p;
    const int l1 = dPL1[p], d1 = 2*l1+1;
    int local = e - dTOFF[p];
    int v = local / d1, i = local % d1;
    const float* xb = &xs[dXOFF[l1]];
    const float* wb = &wls[p*256];
    float s = 0.f;
    for (int u = 0; u < NMUL; ++u) s += xb[u*d1 + i] * wb[u*16 + v];
    ts[e] = s;
  }
  __syncthreads();

  if (tid < KPAD){
    // padded f16 feature copy (B-operand of the pair GEMM)
    Xbuf[(size_t)n*KPAD + tid] = (tid < 144) ? (_Float16)xs[tid] : (_Float16)0.f;

    float hout[9];
    for (int q = 0; q < 9; ++q) hout[q] = 0.f;
    const bool valid = (tid < 144);
    if (valid){
      const int c = tid;
      int l2c, v, j;
      if (c < 16)      { l2c = 0; v = c;           j = 0; }
      else if (c < 64) { l2c = 1; v = (c-16)/3;    j = (c-16)%3; }
      else             { l2c = 2; v = (c-64)/5;    j = (c-64)%5; }
      float K[9];
      for (int q = 0; q < 9; ++q) K[q] = 0.f;
      for (int p = 0; p < NPATH; ++p){
        if (dPL2[p] != l2c) continue;
        const int l1 = dPL1[p], lo = dPLO[p];
        const int d1 = 2*l1+1, d2 = 2*l2c+1, d3 = 2*lo+1;
        const float al = dALPHA[lo];
        const float* tb = &ts[dTOFF[p] + v*d1];
        const float* wb = &w3s[dW3OFF[p] + j*d3];
        const int ko = dKOFF[lo];
        for (int i = 0; i < d1; ++i){
          float tv = al * tb[i];
          for (int k = 0; k < d3; ++k) K[ko+k] += tv * wb[i*d2*d3 + k];
        }
      }
      // fold QCART: QCART[koff(l)+m, a, b] = sqrt(2l+1) * w3j(1,1,l)[a,b,m]
      const int   QP[3] = {1, 5, 11};
      const float sq[3] = {1.f, 1.7320508075688772f, 2.2360679774997896f};
      for (int ab = 0; ab < 9; ++ab){
        float h = 0.f;
        for (int l = 0; l < 3; ++l){
          const int d3 = 2*l+1;
          const float* wb = &w3s[dW3OFF[QP[l]] + ab*d3];
          const int ko = dKOFF[l];
          for (int m = 0; m < d3; ++m) h += sq[l] * wb[m] * K[ko+m];
        }
        hout[ab] = h;
      }
    }
    _Float16* Hr = Hbuf + (size_t)n * (HROWS * KPAD);
    for (int m = 0; m < HROWS; ++m){
      float val = (valid && m < 9) ? hout[m] : 0.f;
      Hr[m*KPAD + tid] = (_Float16)val;
    }
  }
}

// ---------------- stage 3: pair GEMM via v_wmma_f32_16x16x32_f16 -------------
union U16 { v16h v; v8h h[2]; };
union UACC { v8f v; float4 f4[2]; };

__global__ void pair_gemm_kernel(const _Float16* __restrict__ Hbuf,
                                 const _Float16* __restrict__ Xbuf,
                                 float* __restrict__ cart){
  const int n1   = blockIdx.x;            // b*64 + a1  (row node)
  const int lane = threadIdx.x & 31;
  const int wave = threadIdx.x >> 5;      // N-tile 0..3 (16 column atoms each)
  const int b    = n1 >> 6;
  const int half = lane >> 4;             // lane group 0/1
  const int lrow = lane & 15;
  const int a2   = wave*16 + lrow;        // column atom handled by this lane

  const _Float16* Ab = Hbuf + (size_t)n1 * (HROWS*KPAD) + (size_t)lrow * KPAD;
  const _Float16* Bb = Xbuf + (size_t)(b*NATOM + a2) * KPAD;

  UACC acc;
  acc.v = (v8f){};
  for (int kt = 0; kt < 5; ++kt){
    // A 16x32 f16: lane row = lrow; K groups {k0+8h .. +7} and {k0+16+8h .. +7}
    U16 a;
    a.h[0] = *(const v8h*)(Ab + kt*32 + half*8);
    a.h[1] = *(const v8h*)(Ab + kt*32 + 16 + half*8);
    // B 32x16 f16: lane col = lrow; K groups {k0+8h .. +7} and {k0+16+8h .. +7}
    U16 bm;
    bm.h[0] = *(const v8h*)(Bb + kt*32 + half*8);
    bm.h[1] = *(const v8h*)(Bb + kt*32 + 16 + half*8);
    acc.v = __builtin_amdgcn_wmma_f32_16x16x32_f16(false, a.v, false, bm.v,
                                                   (short)0, acc.v, false, false);
  }
  // D layout: lane gives N = lrow, VGPR r gives M = r + 8*half.
  // With CSTRIDE=16, each lane's 8 rows are 8 consecutive floats -> two b128
  // stores, no divergence. Rows 9..15 are exactly zero (H rows are zeroed).
  const size_t e = (size_t)n1 * NATOM + a2;   // pair index (b,a1,a2)
  float* dst = cart + e*CSTRIDE + half*8;
  *(float4*)(dst)     = acc.f4[0];
  *(float4*)(dst + 4) = acc.f4[1];
}

// ---------------- stage 4: Hessian symmetrization ----------------------------
__global__ void sym_kernel(const float* __restrict__ cart, float* __restrict__ out){
  const int e = blockIdx.x * blockDim.x + threadIdx.x;
  if (e >= NPAIRS) return;
  const int b  = e >> 12;
  const int a1 = (e >> 6) & 63;
  const int a2 = e & 63;
  const size_t eT = ((size_t)(b*NATOM + a2)) * NATOM + a1;
  for (int i = 0; i < 3; ++i)
    for (int j = 0; j < 3; ++j)
      out[(size_t)e*9 + i*3 + j] =
          0.5f * (cart[(size_t)e*CSTRIDE + i*3 + j] + cart[eT*CSTRIDE + j*3 + i]);
}

extern "C" void kernel_launch(void* const* d_in, const int* in_sizes, int n_in,
                              void* d_out, int out_size, void* d_ws, size_t ws_size,
                              hipStream_t stream){
  (void)in_sizes; (void)n_in; (void)out_size; (void)ws_size;
  const float* node_feats = (const float*)d_in[0];
  // d_in[1] (layout) is the dense per-config (a1, a2) meshgrid; indexing is implicit.
  const float* tp_weights = (const float*)d_in[2];

  char* ws = (char*)d_ws;
  float*    w3   = (float*)(ws);                                        // 615 f32
  _Float16* Hbuf = (_Float16*)(ws + 4096);                              // 4096*16*160 f16
  _Float16* Xbuf = (_Float16*)(ws + 4096 + (size_t)NNODES*HROWS*KPAD*2);// 4096*160 f16
  float*    cart = (float*)(ws + 4096 + (size_t)NNODES*HROWS*KPAD*2
                                       + (size_t)NNODES*KPAD*2);        // NPAIRS*16 f32

  hipLaunchKernelGGL(w3j_kernel, dim3(1), dim3(640), 0, stream, w3);
  hipLaunchKernelGGL(build_h_kernel, dim3(NNODES), dim3(256), 0, stream,
                     node_feats, tp_weights, w3, Hbuf, Xbuf);
  hipLaunchKernelGGL(pair_gemm_kernel, dim3(NNODES), dim3(128), 0, stream,
                     Hbuf, Xbuf, cart);
  hipLaunchKernelGGL(sym_kernel, dim3((NPAIRS + 255)/256), dim3(256), 0, stream,
                     cart, (float*)d_out);
}
